// Roost_11510512354059
// MI455X (gfx1250) — compile-verified, matrix operand order checked by reading, not run on hardware
//
#include <hip/hip_runtime.h>
#include <hip/hip_bf16.h>
#include <stdint.h>

#define H   128
#define TWOH 256
#define EPSB 1e-5f

typedef __attribute__((ext_vector_type(16))) _Float16 v16h;
typedef __attribute__((ext_vector_type(8)))  _Float16 v8h;
typedef __attribute__((ext_vector_type(4)))  _Float16 v4h;
typedef __attribute__((ext_vector_type(8)))  float    v8f;
typedef __attribute__((ext_vector_type(4)))  float    v4f;

// ---------------- zero scratch ----------------
__global__ void k_zero(float* __restrict__ p, long n) {
  long i  = (long)blockIdx.x * blockDim.x + threadIdx.x;
  long st = (long)gridDim.x * blockDim.x;
  for (; i < n; i += st) p[i] = 0.0f;
}

// ---------------- main WMMA edge-GEMM ----------------
// block = 128 threads = 4 waves; each wave computes one 16-edge tile:
//   z[16 x 128] = hihj[16 x 256] @ Wg[256 x 128]   (8 N-tiles x 8 K-steps of wmma f16)
//   l[16]       = hihj[16 x 256] @ [Wf | 0..0]     (1 extra N-tile, 8 more wmma)
// and accumulates BN sums for both.
__global__ __launch_bounds__(128) void k_gemm(
    const float* __restrict__ x, const int* __restrict__ src, const int* __restrict__ dst,
    const float* __restrict__ Wf, const float* __restrict__ bf,
    const float* __restrict__ Wg, const float* __restrict__ bg,
    float* __restrict__ l_out, _Float16* __restrict__ z_out,
    float* __restrict__ stats, int E)
{
  __shared__ __attribute__((aligned(16))) _Float16 Bs[H * TWOH];      // 64 KB, col-major per out ch
  __shared__ __attribute__((aligned(16))) _Float16 Bf[16 * TWOH];     //  8 KB, col0 = Wf, rest 0
  __shared__ __attribute__((aligned(16))) _Float16 As[4][16 * TWOH];  // 32 KB, per-wave edge tile
  __shared__ float zsum[H], zsq[H];
  __shared__ float lstat[2];

  const int tid  = threadIdx.x;
  const int lane = tid & 31;
  const int w    = tid >> 5;

  // Wg [4][256][32] -> Bs[j*256 + c] f16, j = head*32 + k (matches einsum/g_gamma flat layout)
  for (int idx = tid; idx < H * TWOH; idx += 128) {
    int j = idx >> 8, c = idx & 255;
    Bs[idx] = (_Float16)Wg[(size_t)(j >> 5) * 8192 + (size_t)c * 32 + (j & 31)];
  }
  // Bf: column 0 holds Wf, columns 1..15 zero
  for (int idx = tid; idx < 16 * TWOH; idx += 128) {
    int n = idx >> 8, c = idx & 255;
    Bf[idx] = (n == 0) ? (_Float16)Wf[c] : (_Float16)0.0f;
  }
  if (tid < H) { zsum[tid] = 0.0f; zsq[tid] = 0.0f; }
  if (tid < 2) lstat[tid] = 0.0f;
  __syncthreads();

  const int ebase    = (blockIdx.x * 4 + w) * 16;
  const int m16      = lane & 15;
  const int half_off = (lane < 16) ? 0 : H;    // lanes 0-15 gather src half, 16-31 dst half
  const int cb       = m16 * 8;

  // stage A tile: 16 edges x 256 cols as f16 in LDS (vectorized gather, rows L2-resident)
  for (int r = 0; r < 16; ++r) {
    int e = ebase + r; if (e >= E) e = E - 1;
    int node = (lane < 16) ? src[e] : dst[e];
    const float* rowp = x + (size_t)node * H;
    v4f f0 = *(const v4f*)(rowp + cb);
    v4f f1 = *(const v4f*)(rowp + cb + 4);
    v8h hh;
#pragma unroll
    for (int i = 0; i < 4; ++i) { hh[i] = (_Float16)f0[i]; hh[4 + i] = (_Float16)f1[i]; }
    *(v8h*)&As[w][r * TWOH + half_off + cb] = hh;
  }
  __syncthreads();

  // A fragments in ISA layout: lane holds row m16; lo-half lanes K={kb..kb+7,kb+16..23},
  // hi-half lanes K={kb+8..15,kb+24..31}  -> two contiguous 16B LDS loads each.
  const int khi = (lane < 16) ? 0 : 8;
  v16h afrag[8];
#pragma unroll
  for (int kk = 0; kk < 8; ++kk) {
    const v8h* p = (const v8h*)&As[w][m16 * TWOH + kk * 32 + khi];
    v8h lo = p[0], hi = p[2];
#pragma unroll
    for (int i = 0; i < 8; ++i) { afrag[kk][i] = lo[i]; afrag[kk][8 + i] = hi[i]; }
  }

  const int rowhi = (lane < 16) ? 0 : 8;   // C/D rows: VGPR r -> M = r + 8*laneHi

  // ---- g: 8 N-tiles of 16 output channels ----
#pragma unroll
  for (int nt = 0; nt < 8; ++nt) {
    const int j = nt * 16 + m16;           // output channel for this lane
    v8f acc = {};
#pragma unroll
    for (int kk = 0; kk < 8; ++kk) {
      const v8h* p = (const v8h*)&Bs[j * TWOH + kk * 32 + khi];
      v8h lo = p[0], hi = p[2];
      v16h bfrag;
#pragma unroll
      for (int i = 0; i < 8; ++i) { bfrag[i] = lo[i]; bfrag[8 + i] = hi[i]; }
      acc = __builtin_amdgcn_wmma_f32_16x16x32_f16(
          false, afrag[kk], false, bfrag, (short)0, acc, false, false);
    }
    float bgj = bg[j];
    float ps = 0.0f, pq = 0.0f;
#pragma unroll
    for (int r = 0; r < 8; ++r) {
      int e = ebase + r + rowhi;
      float v = acc[r] + bgj;
      if (e < E) {
        z_out[(size_t)e * H + j] = (_Float16)v;
        ps += v; pq += v * v;
      }
    }
    atomicAdd(&zsum[j], ps);
    atomicAdd(&zsq[j],  pq);
  }

  // ---- f: one N-tile against [Wf | 0...0]; logit lives in column n==0 ----
  {
    v8f acc = {};
#pragma unroll
    for (int kk = 0; kk < 8; ++kk) {
      const v8h* p = (const v8h*)&Bf[m16 * TWOH + kk * 32 + khi];
      v8h lo = p[0], hi = p[2];
      v16h bfrag;
#pragma unroll
      for (int i = 0; i < 8; ++i) { bfrag[i] = lo[i]; bfrag[8 + i] = hi[i]; }
      acc = __builtin_amdgcn_wmma_f32_16x16x32_f16(
          false, afrag[kk], false, bfrag, (short)0, acc, false, false);
    }
    if (m16 == 0) {                        // lanes 0 and 16 hold column 0 (rows 0-7 / 8-15)
      float bf0 = bf[0];
      float ps = 0.0f, pq = 0.0f;
#pragma unroll
      for (int r = 0; r < 8; ++r) {
        int e = ebase + r + rowhi;
        float lv = acc[r] + bf0;
        if (e < E) {
          l_out[e] = lv;
          ps += lv; pq += lv * lv;
        }
      }
      atomicAdd(&lstat[0], ps);
      atomicAdd(&lstat[1], pq);
    }
  }
  __syncthreads();

  if (tid < H) {
    atomicAdd(&stats[2 + tid],   zsum[tid]);
    atomicAdd(&stats[130 + tid], zsq[tid]);
  }
  if (tid == 0) { atomicAdd(&stats[0], lstat[0]); atomicAdd(&stats[1], lstat[1]); }
}

// ---------------- finalize edge BN stats ----------------
__global__ void k_finalize(float* __restrict__ stats, int E) {
  int tid = threadIdx.x;
  float invE = 1.0f / (float)E;
  if (tid < H) {
    float m = stats[2 + tid] * invE;
    float v = stats[130 + tid] * invE - m * m;
    stats[260 + tid] = m;
    stats[388 + tid] = rsqrtf(v + EPSB);
  } else if (tid == H) {
    float m = stats[0] * invE;
    float v = stats[1] * invE - m * m;
    stats[258] = m;
    stats[259] = rsqrtf(v + EPSB);
  }
}

// ---------------- f path: exp(SiLU(BN(l))), weighted dst-sum ----------------
__global__ void k_edge(const float* __restrict__ l_ws, const float* __restrict__ weight,
                       const int* __restrict__ src, const int* __restrict__ dst,
                       const float* __restrict__ f_gamma, const float* __restrict__ f_beta,
                       const float* __restrict__ stats,
                       float* __restrict__ we_ws, float* __restrict__ sumw, int E) {
  int e = blockIdx.x * blockDim.x + threadIdx.x;
  if (e >= E) return;
  float t = (l_ws[e] - stats[258]) * stats[259] * f_gamma[0] + f_beta[0];
  float s = t / (1.0f + __expf(-t));          // SiLU
  float wv = weight[src[e]] * __expf(s);
  we_ws[e] = wv;
  atomicAdd(&sumw[dst[e]], wv);
}

// ---------------- message: a * SiLU(BN(z)) scattered into h ----------------
__global__ void k_msg(const _Float16* __restrict__ z_ws, const float* __restrict__ we_ws,
                      const float* __restrict__ sumw, const int* __restrict__ dst,
                      const float* __restrict__ g_gamma, const float* __restrict__ g_beta,
                      const float* __restrict__ stats, float* __restrict__ h_ws, int E) {
  long idx = (long)blockIdx.x * blockDim.x + threadIdx.x;   // one thread = (edge, 4 channels)
  long total = (long)E * 32;
  if (idx >= total) return;
  int e  = (int)(idx >> 5);
  int j0 = ((int)idx & 31) * 4;
  int d  = dst[e];
  float a = we_ws[e] / sumw[d];
  v4h zv = *(const v4h*)&z_ws[(size_t)e * H + j0];
#pragma unroll
  for (int i = 0; i < 4; ++i) {
    int j = j0 + i;
    float zn = ((float)zv[i] - stats[260 + j]) * stats[388 + j] * g_gamma[j] + g_beta[j];
    float s = zn / (1.0f + __expf(-zn));
    atomicAdd(&h_ws[(size_t)d * H + j], a * s);
  }
}

// ---------------- node BN stats: one block per channel ----------------
__global__ void k_nodestats(const float* __restrict__ h_ws, float* __restrict__ stats, int N) {
  __shared__ float s1[256], s2[256];
  int ch = blockIdx.x, tid = threadIdx.x;
  float ps = 0.0f, pq = 0.0f;
  for (int n = tid; n < N; n += 256) {
    float v = h_ws[(size_t)n * H + ch];
    ps += v; pq += v * v;
  }
  s1[tid] = ps; s2[tid] = pq;
  __syncthreads();
  for (int off = 128; off > 0; off >>= 1) {
    if (tid < off) { s1[tid] += s1[tid + off]; s2[tid] += s2[tid + off]; }
    __syncthreads();
  }
  if (tid == 0) {
    float invN = 1.0f / (float)N;
    float m = s1[0] * invN;
    float v = s2[0] * invN - m * m;
    stats[516 + ch] = m;
    stats[644 + ch] = rsqrtf(v + EPSB);
  }
}

// ---------------- final BN + residual ----------------
__global__ void k_out(const float* __restrict__ h_ws, const float* __restrict__ x,
                      const float* __restrict__ n_gamma, const float* __restrict__ n_beta,
                      const float* __restrict__ stats, float* __restrict__ out, int N) {
  long idx = (long)blockIdx.x * blockDim.x + threadIdx.x;
  if (idx >= (long)N * H) return;
  int j = (int)(idx & (H - 1));
  out[idx] = (h_ws[idx] - stats[516 + j]) * stats[644 + j] * n_gamma[j] + n_beta[j] + x[idx];
}

extern "C" void kernel_launch(void* const* d_in, const int* in_sizes, int n_in,
                              void* d_out, int out_size, void* d_ws, size_t ws_size,
                              hipStream_t stream) {
  const float* x       = (const float*)d_in[0];
  const float* weight  = (const float*)d_in[1];
  const int*   src     = (const int*)  d_in[2];
  const int*   dst     = (const int*)  d_in[3];
  const float* Wf      = (const float*)d_in[4];
  const float* bf      = (const float*)d_in[5];
  const float* f_gamma = (const float*)d_in[6];
  const float* f_beta  = (const float*)d_in[7];
  const float* Wg      = (const float*)d_in[8];
  const float* bg      = (const float*)d_in[9];
  const float* g_gamma = (const float*)d_in[10];
  const float* g_beta  = (const float*)d_in[11];
  const float* n_gamma = (const float*)d_in[12];
  const float* n_beta  = (const float*)d_in[13];

  const int N = in_sizes[1];   // weight is [N,1]
  const int E = in_sizes[2];   // src is [E]

  // workspace layout (floats): l[E] | we[E] | sumw[N] | h[N*128] | stats[1024] | z[E*128] f16
  float* wsf   = (float*)d_ws;
  float* l_ws  = wsf;
  float* we_ws = wsf + (size_t)E;
  float* sumw  = wsf + 2 * (size_t)E;
  float* h_ws  = sumw + (size_t)N;
  float* stats = h_ws + (size_t)N * H;
  uintptr_t zp = ((uintptr_t)(stats + 1024) + 63) & ~(uintptr_t)63;
  _Float16* z_ws = (_Float16*)zp;

  // zero sumw + h + stats (contiguous from sumw)
  long zn = (long)N + (long)N * H + 1024;
  k_zero<<<1024, 256, 0, stream>>>(sumw, zn);

  int gblk = (E + 63) / 64;                    // 64 edges per block (4 waves x 16)
  k_gemm<<<gblk, 128, 0, stream>>>(x, src, dst, Wf, bf, Wg, bg, l_ws, z_ws, stats, E);

  k_finalize<<<1, 256, 0, stream>>>(stats, E);

  k_edge<<<(E + 255) / 256, 256, 0, stream>>>(l_ws, weight, src, dst, f_gamma, f_beta,
                                              stats, we_ws, sumw, E);

  long mt = (long)E * 32;
  k_msg<<<(int)((mt + 255) / 256), 256, 0, stream>>>(z_ws, we_ws, sumw, dst,
                                                     g_gamma, g_beta, stats, h_ws, E);

  k_nodestats<<<H, 256, 0, stream>>>(h_ws, stats, N);

  k_out<<<(int)(((long)N * H + 255) / 256), 256, 0, stream>>>(h_ws, x, n_gamma, n_beta,
                                                              stats, (float*)d_out, N);
}
